// GCN_Net_57191784513886
// MI455X (gfx1250) — matrix operand order repaired and checked by reading, not compiled, not run on hardware
//
#include <hip/hip_runtime.h>

typedef __attribute__((ext_vector_type(16))) _Float16 v16h;
typedef __attribute__((ext_vector_type(8)))  _Float16 v8h;
typedef __attribute__((ext_vector_type(8)))  float    v8f;

#define NN 50000
#define NE 800000

// ---------------- degrees ----------------
__global__ void degree_kernel(const int* __restrict__ src, const int* __restrict__ dst,
                              float* __restrict__ deg_out, float* __restrict__ deg_in) {
    int e = blockIdx.x * blockDim.x + threadIdx.x;
    if (e < NE) {
        unsafeAtomicAdd(&deg_out[src[e]], 1.0f);
        unsafeAtomicAdd(&deg_in [dst[e]], 1.0f);
    }
}

__global__ void rsqrt_kernel(float* __restrict__ deg_out, float* __restrict__ deg_in) {
    int i = blockIdx.x * blockDim.x + threadIdx.x;
    if (i < NN) {
        deg_out[i] = rsqrtf(fmaxf(deg_out[i], 1.0f));
        deg_in [i] = rsqrtf(fmaxf(deg_in [i], 1.0f));
    }
}

// ---- pack W (KxN row-major f32) into WT (NxK row-major f16) for B-fragment loads ----
__global__ void pack_w(const float* __restrict__ W, _Float16* __restrict__ WT, int K, int N) {
    int idx = blockIdx.x * blockDim.x + threadIdx.x;
    if (idx < K * N) {
        int n = idx / K, k = idx % K;
        WT[idx] = (_Float16)W[k * N + n];
    }
}

// ---- Xh[i,k] = f16( feat[i,k] * rsq_out[i] ), 4 elems/thread ----
__global__ void scale_cvt(const float* __restrict__ X, const float* __restrict__ s,
                          _Float16* __restrict__ Xh, int F, int total4) {
    int idx = blockIdx.x * blockDim.x + threadIdx.x;
    if (idx < total4) {
        int base = idx * 4;
        float sc = s[base / F];
        float4 x = ((const float4*)X)[idx];
        Xh[base + 0] = (_Float16)(x.x * sc);
        Xh[base + 1] = (_Float16)(x.y * sc);
        Xh[base + 2] = (_Float16)(x.z * sc);
        Xh[base + 3] = (_Float16)(x.w * sc);
    }
}

// ---------------- WMMA GEMM: C[M,N] = A[M,K] * BT[N,K]^T ----------------
// blockDim = (32, N/16); gridDim.x = M/16. One wave computes one 16x16 C tile.
__global__ void wmma_gemm(const _Float16* __restrict__ A, const _Float16* __restrict__ BT,
                          float* __restrict__ C, int K, int N) {
    const int lane = threadIdx.x;
    const int hs   = lane >> 4;     // half-wave select
    const int l15  = lane & 15;

    const int arow = blockIdx.x * 16 + l15;       // A row for this lane
    const int bcol = threadIdx.y * 16 + l15;      // B column for this lane

    // A fragment: lanes 0-15 hold K {k0..k0+15}; lanes 16-31 K {k0+16..k0+31}
    const _Float16* ap = A  + (size_t)arow * K + hs * 16;
    // B fragment: lane = N, lanes 0-15 K {k0..k0+15}, lanes 16-31 K {k0+16..k0+31}
    const _Float16* bp = BT + (size_t)bcol * K + hs * 16;

    v8f c = {};
    for (int k0 = 0; k0 < K; k0 += 32) {
        union { v16h v; v8h h[2]; } a;
        a.h[0] = *(const v8h*)(ap + k0);
        a.h[1] = *(const v8h*)(ap + k0 + 8);
        v16h b = *(const v16h*)(bp + k0);
        c = __builtin_amdgcn_wmma_f32_16x16x32_f16(false, a.v, false, b,
                                                   (short)0, c, false, false);
    }
    // C/D layout: lane l -> col = l&15, rows = hs*8 + j
    const int mbase = blockIdx.x * 16 + hs * 8;
    float* cp = C + (size_t)mbase * N + threadIdx.y * 16 + l15;
#pragma unroll
    for (int j = 0; j < 8; ++j) cp[(size_t)j * N] = c[j];
}

// ---------------- edge aggregation: Msum[dst] += H[src], TPE threads/edge, 4 f32/thread ----
template <int TPE>
__global__ void aggregate(const float* __restrict__ H, const int* __restrict__ src,
                          const int* __restrict__ dst, float* __restrict__ Msum, int F) {
    int idx = blockIdx.x * blockDim.x + threadIdx.x;
    int e = idx / TPE;
    int f = (idx % TPE) * 4;
    if (e < NE) {
        int s = src[e], d = dst[e];
        float4 v = *(const float4*)(H + (size_t)s * F + f);
        float* p = Msum + (size_t)d * F + f;
        unsafeAtomicAdd(p + 0, v.x);
        unsafeAtomicAdd(p + 1, v.y);
        unsafeAtomicAdd(p + 2, v.z);
        unsafeAtomicAdd(p + 3, v.w);
    }
}

// ---- X2h = f16( relu(Msum * rsq_in + b1) * rsq_out ), 4 elems/thread ----
__global__ void relu_bias_scale(const float* __restrict__ Msum, const float* __restrict__ rin,
                                const float* __restrict__ rout, const float* __restrict__ bias,
                                _Float16* __restrict__ Xh, int F, int total4) {
    int idx = blockIdx.x * blockDim.x + threadIdx.x;
    if (idx < total4) {
        int base = idx * 4;
        int row = base / F, f = base % F;
        float ri = rin[row], ro = rout[row];
        float4 m = ((const float4*)Msum)[idx];
        Xh[base + 0] = (_Float16)(fmaxf(m.x * ri + bias[f + 0], 0.0f) * ro);
        Xh[base + 1] = (_Float16)(fmaxf(m.y * ri + bias[f + 1], 0.0f) * ro);
        Xh[base + 2] = (_Float16)(fmaxf(m.z * ri + bias[f + 2], 0.0f) * ro);
        Xh[base + 3] = (_Float16)(fmaxf(m.w * ri + bias[f + 3], 0.0f) * ro);
    }
}

// ---- out = log_softmax(M2 * rsq_in + b2), one wave32 = one node (32 classes) ----
__global__ void logsoftmax_kernel(const float* __restrict__ M2, const float* __restrict__ rin,
                                  const float* __restrict__ b2, float* __restrict__ out) {
    int t = blockIdx.x * blockDim.x + threadIdx.x;
    int node = t >> 5;
    int c = t & 31;
    if (node < NN) {
        float v = M2[(size_t)node * 32 + c] * rin[node] + b2[c];
        float mx = v;
        for (int off = 16; off; off >>= 1) mx = fmaxf(mx, __shfl_xor(mx, off));
        float ex = expf(v - mx);
        float s = ex;
        for (int off = 16; off; off >>= 1) s += __shfl_xor(s, off);
        out[(size_t)node * 32 + c] = v - mx - logf(s);
    }
}

extern "C" void kernel_launch(void* const* d_in, const int* in_sizes, int n_in,
                              void* d_out, int out_size, void* d_ws, size_t ws_size,
                              hipStream_t stream) {
    const float* feat = (const float*)d_in[0];
    const int*   src  = (const int*)  d_in[1];
    const int*   dst  = (const int*)  d_in[2];
    const float* W1   = (const float*)d_in[3];
    const float* b1   = (const float*)d_in[4];
    const float* W2   = (const float*)d_in[5];
    const float* b2   = (const float*)d_in[6];
    float* out = (float*)d_out;

    // workspace carve-up (256B aligned)
    size_t off = 0;
    auto carve = [&](size_t bytes) -> void* {
        void* p = (char*)d_ws + off;
        off += (bytes + 255) & ~(size_t)255;
        return p;
    };
    float*     deg_out = (float*)    carve((size_t)NN * 4);
    float*     deg_in  = (float*)    carve((size_t)NN * 4);
    _Float16*  W1T     = (_Float16*) carve((size_t)128 * 256 * 2);
    _Float16*  W2T     = (_Float16*) carve((size_t)32 * 128 * 2);
    _Float16*  Xh      = (_Float16*) carve((size_t)NN * 256 * 2);  // reused as X2h
    float*     H       = (float*)    carve((size_t)NN * 128 * 4);  // reused as H2
    float*     Msum    = (float*)    carve((size_t)NN * 128 * 4);  // reused as M2

    // degrees -> rsqrt of clipped degrees
    hipMemsetAsync(deg_out, 0, (size_t)NN * 4, stream);
    hipMemsetAsync(deg_in,  0, (size_t)NN * 4, stream);
    degree_kernel<<<(NE + 255) / 256, 256, 0, stream>>>(src, dst, deg_out, deg_in);
    rsqrt_kernel<<<(NN + 255) / 256, 256, 0, stream>>>(deg_out, deg_in);

    // pack weights to f16, transposed (B operand wants contiguous-K per column)
    pack_w<<<(256 * 128 + 255) / 256, 256, 0, stream>>>(W1, W1T, 256, 128);
    pack_w<<<(128 * 32 + 255) / 256, 256, 0, stream>>>(W2, W2T, 128, 32);

    // layer 1
    scale_cvt<<<(NN * 64 + 255) / 256, 256, 0, stream>>>(feat, deg_out, Xh, 256, NN * 64);
    wmma_gemm<<<dim3(NN / 16), dim3(32, 8), 0, stream>>>(Xh, W1T, H, 256, 128);
    hipMemsetAsync(Msum, 0, (size_t)NN * 128 * 4, stream);
    aggregate<32><<<(NE * 32) / 256, 256, 0, stream>>>(H, src, dst, Msum, 128);
    relu_bias_scale<<<(NN * 32 + 255) / 256, 256, 0, stream>>>(Msum, deg_in, deg_out, b1,
                                                               Xh, 128, NN * 32);

    // layer 2
    wmma_gemm<<<dim3(NN / 16), dim3(32, 2), 0, stream>>>(Xh, W2T, H, 128, 32);
    hipMemsetAsync(Msum, 0, (size_t)NN * 32 * 4, stream);
    aggregate<8><<<(NE * 8) / 256, 256, 0, stream>>>(H, src, dst, Msum, 32);

    // normalize + bias + log_softmax
    logsoftmax_kernel<<<(NN * 32 + 255) / 256, 256, 0, stream>>>(Msum, deg_in, b2, out);
}
